// PolynomialSketch_71253507441243
// MI455X (gfx1250) — matrix operand order repaired and checked by reading, not compiled, not run on hardware
//
#include <hip/hip_runtime.h>

typedef __attribute__((ext_vector_type(16))) __bf16 v16bf;
typedef __attribute__((ext_vector_type(8)))  float  v8f;
typedef __attribute__((ext_vector_type(4)))  float  v4f;
typedef __attribute__((ext_vector_type(4)))  __bf16 v4bf;
typedef __attribute__((ext_vector_type(4)))  int    v4i;

#define BATCH  16384
#define D_IN   1024
#define D_FEAT 4096
#define DEG    3

// GEMM tiling
#define BM 128
#define BN 64
#define KT 32
#define ROW_BYTES 80   // 40 bf16 per LDS row (32 + 8 pad), 16B-aligned stride

#if __has_builtin(__builtin_amdgcn_global_load_async_to_lds_b128) && \
    __has_builtin(__builtin_amdgcn_s_wait_asynccnt)
#define HAS_ASYNC 1
#else
#define HAS_ASYNC 0
#endif

#if HAS_ASYNC
// DMA 16 bytes global -> LDS, tracked by ASYNCcnt (no VGPR round-trip).
// Builtin is declared with generic v4i* params on this toolchain; the backend
// infers global vs LDS address spaces from pointer provenance.
#define ASYNC_COPY16(gsrc, ldst)                                        \
    __builtin_amdgcn_global_load_async_to_lds_b128(                     \
        (v4i*)(gsrc), (v4i*)(ldst), 0, 0)
#define WAIT_ASYNC() __builtin_amdgcn_s_wait_asynccnt(0)
#else
#define ASYNC_COPY16(gsrc, ldst)                                        \
    (*reinterpret_cast<uint4*>(ldst) =                                  \
         *reinterpret_cast<const uint4*>(gsrc))
#define WAIT_ASYNC() ((void)0)
#endif

union Frag {
    v16bf v;
    uint4 q[2];
};

// ---------------------------------------------------------------------------
// Pass 1: xs = bf16(x * exp(-log_lengthscale)), row-major [BATCH, D_IN]
// ---------------------------------------------------------------------------
__global__ void xs_convert_kernel(const float* __restrict__ x,
                                  const float* __restrict__ log_ls,
                                  __bf16* __restrict__ xs) {
    const float s = __expf(-log_ls[0]);
    size_t i = ((size_t)blockIdx.x * blockDim.x + threadIdx.x) * 4;
    v4f v = *reinterpret_cast<const v4f*>(x + i);
    v *= s;
    *reinterpret_cast<v4bf*>(xs + i) = __builtin_convertvector(v, v4bf);
}

// ---------------------------------------------------------------------------
// Pass 2: Wt[k][o][i] = bf16(W[k][i][o])  (K-contiguous for coalesced staging)
// 32x32 LDS-tiled transpose, block (32,8), 4 rows per thread
// ---------------------------------------------------------------------------
__global__ void wt_convert_kernel(const float* __restrict__ W,
                                  __bf16* __restrict__ Wt) {
    __shared__ float tile[32][33];
    const int k  = blockIdx.z;
    const int o0 = blockIdx.x * 32;
    const int i0 = blockIdx.y * 32;
    const int tx = threadIdx.x, ty = threadIdx.y;
#pragma unroll
    for (int r = 0; r < 4; ++r) {
        int i = i0 + ty + r * 8;
        tile[ty + r * 8][tx] = W[((size_t)k * D_IN + i) * D_FEAT + (o0 + tx)];
    }
    __syncthreads();
#pragma unroll
    for (int r = 0; r < 4; ++r) {
        int o = o0 + ty + r * 8;
        Wt[((size_t)k * D_FEAT + o) * D_IN + (i0 + tx)] =
            (__bf16)tile[tx][ty + r * 8];
    }
}

// ---------------------------------------------------------------------------
// Pass 3: fused triple-GEMM + elementwise-product epilogue.
// out[b][o] = alpha * prod_k ( sum_i xs[b][i] * Wt[k][o][i] )
// Block tile: BM=128 x BN=64, 8 wave32s (4x2), each wave 32x32 = 2x2 WMMA
// tiles x 3 degrees. bf16 WMMA, K-step 32, double-buffered LDS with async
// global->LDS DMA overlapping the WMMAs.
// ---------------------------------------------------------------------------
__global__ __launch_bounds__(256)
void sketch_gemm_kernel(const __bf16* __restrict__ xs,
                        const __bf16* __restrict__ Wt,
                        const float* __restrict__ log_var,
                        float* __restrict__ out) {
    __shared__ alignas(16) unsigned char sA[2][BM * ROW_BYTES];        // 2x10 KB
    __shared__ alignas(16) unsigned char sB[2][DEG][BN * ROW_BYTES];   // 2x15 KB

    const int tid  = threadIdx.x;
    const int lane = tid & 31;
    const int wave = tid >> 5;
    const int wm   = wave & 3;   // wave row (4)
    const int wn   = wave >> 2;  // wave col (2)
    const int lh   = lane >> 4;  // half-wave
    const int lm   = lane & 15;

    const int rowBase = blockIdx.y * BM;
    const int colBase = blockIdx.x * BN;

    // staging roles
    const int aRow   = tid >> 1;  // 0..127, 16 bf16 per thread
    const int aHalf  = tid & 1;
    const int bCol   = tid >> 2;  // 0..63, 8 bf16 per thread per degree
    const int bChunk = tid & 3;

    const unsigned short* aG =
        (const unsigned short*)xs + (size_t)(rowBase + aRow) * D_IN + aHalf * 16;
    const unsigned short* bG0 =
        (const unsigned short*)Wt +
        ((size_t)0 * D_FEAT + colBase + bCol) * D_IN + bChunk * 8;
    const unsigned short* bG1 = bG0 + (size_t)1 * D_FEAT * D_IN;
    const unsigned short* bG2 = bG0 + (size_t)2 * D_FEAT * D_IN;

    const int aOff = aRow * ROW_BYTES + aHalf * 32;
    const int bOff = bCol * ROW_BYTES + bChunk * 16;

#define STAGE(bi, kk)                                                   \
    do {                                                                \
        ASYNC_COPY16(aG + (kk),      &sA[bi][aOff]);                    \
        ASYNC_COPY16(aG + (kk) + 8,  &sA[bi][aOff + 16]);               \
        ASYNC_COPY16(bG0 + (kk),     &sB[bi][0][bOff]);                 \
        ASYNC_COPY16(bG1 + (kk),     &sB[bi][1][bOff]);                 \
        ASYNC_COPY16(bG2 + (kk),     &sB[bi][2][bOff]);                 \
    } while (0)

    v8f acc[DEG][2][2];
#pragma unroll
    for (int d = 0; d < DEG; ++d)
#pragma unroll
        for (int mt = 0; mt < 2; ++mt)
#pragma unroll
            for (int nt = 0; nt < 2; ++nt)
                acc[d][mt][nt] = (v8f)0.0f;

    // prologue: fill buffer 0
    STAGE(0, 0);
    WAIT_ASYNC();
    __syncthreads();

    int buf = 0;
    for (int kt = 0; kt < D_IN; kt += KT) {
        const int nb = buf ^ 1;
        // kick off DMA of the next tile while we compute on the current one
        if (kt + KT < D_IN) STAGE(nb, kt + KT);

        // ---- A fragments (ISA 16-bit A 16x32 layout):
        //      chunk0 = K[lh*8 .. +7], chunk1 = K[16+lh*8 .. +7]
        Frag a[2];
#pragma unroll
        for (int mt = 0; mt < 2; ++mt) {
            const int row = wm * 32 + mt * 16 + lm;
            const unsigned char* p = &sA[buf][row * ROW_BYTES + lh * 16];
            a[mt].q[0] = *reinterpret_cast<const uint4*>(p);
            a[mt].q[1] = *reinterpret_cast<const uint4*>(p + 32);
        }

#pragma unroll
        for (int d = 0; d < DEG; ++d) {
            // ---- B fragments (ISA 16-bit B 32x16 layout):
            //      col = lm, chunk0 = K[lh*8 .. +7], chunk1 = K[16+lh*8 .. +7]
            Frag b[2];
#pragma unroll
            for (int nt = 0; nt < 2; ++nt) {
                const int col = wn * 32 + nt * 16 + lm;
                const unsigned char* p = &sB[buf][d][col * ROW_BYTES + lh * 16];
                b[nt].q[0] = *reinterpret_cast<const uint4*>(p);
                b[nt].q[1] = *reinterpret_cast<const uint4*>(p + 32);
            }
#pragma unroll
            for (int mt = 0; mt < 2; ++mt)
#pragma unroll
                for (int nt = 0; nt < 2; ++nt)
                    acc[d][mt][nt] = __builtin_amdgcn_wmma_f32_16x16x32_bf16(
                        false, a[mt].v, false, b[nt].v, (short)0,
                        acc[d][mt][nt], false, false);
        }

        // next buffer's DMA must be complete (own wave), then block-wide sync
        WAIT_ASYNC();
        __syncthreads();
        buf = nb;
    }

    // ---- epilogue: elementwise product over degree + scale; NT stores so the
    //      write-once output doesn't evict the L2-resident bf16 operands ----
    const float alpha = 0.015625f /* 1/sqrt(4096) */ * __expf(0.5f * log_var[0]);
#pragma unroll
    for (int mt = 0; mt < 2; ++mt)
#pragma unroll
        for (int nt = 0; nt < 2; ++nt) {
            v8f r = acc[0][mt][nt] * acc[1][mt][nt] * acc[2][mt][nt] * alpha;
            const int col   = colBase + wn * 32 + nt * 16 + lm;
            const int rbase = rowBase + wm * 32 + mt * 16 + lh * 8;
#pragma unroll
            for (int v = 0; v < 8; ++v)
                __builtin_nontemporal_store(
                    r[v], &out[(size_t)(rbase + v) * D_FEAT + col]);
        }
#undef STAGE
}

// ---------------------------------------------------------------------------
extern "C" void kernel_launch(void* const* d_in, const int* in_sizes, int n_in,
                              void* d_out, int out_size, void* d_ws,
                              size_t ws_size, hipStream_t stream) {
    const float* x       = (const float*)d_in[0];
    const float* W       = (const float*)d_in[1];
    const float* log_ls  = (const float*)d_in[2];
    const float* log_var = (const float*)d_in[3];
    float* out = (float*)d_out;

    __bf16* xs = (__bf16*)d_ws;                       // 16384*1024*2 = 32 MB
    __bf16* Wt = xs + (size_t)BATCH * D_IN;           // 3*4096*1024*2 = 24 MB

    // Pass 1: scale + convert x -> bf16
    xs_convert_kernel<<<(BATCH * D_IN) / (256 * 4), 256, 0, stream>>>(
        x, log_ls, xs);

    // Pass 2: transpose + convert W -> bf16 [deg][o][i]
    wt_convert_kernel<<<dim3(D_FEAT / 32, D_IN / 32, DEG), dim3(32, 8), 0,
                        stream>>>(W, Wt);

    // Pass 3: fused triple GEMM + product epilogue
    sketch_gemm_kernel<<<dim3(D_FEAT / BN, BATCH / BM), 256, 0, stream>>>(
        xs, Wt, log_var, out);
}
